// EKANLayer_26225070309753
// MI455X (gfx1250) — matrix operand order repaired and checked by reading, not compiled, not run on hardware
//
#include <hip/hip_runtime.h>
#include <math.h>

// ---------------------------------------------------------------------------
// Types for CDNA5 WMMA (wave32). v_wmma_f32_16x16x32_bf16: A/B = v16bf, C/D = v8f
// ---------------------------------------------------------------------------
typedef __attribute__((ext_vector_type(16))) __bf16 v16bf;
typedef __attribute__((ext_vector_type(8)))  __bf16 bf16x8;
typedef __attribute__((ext_vector_type(8)))  float  v8f;

#define BATCH    8192
#define GATED_IN 576
#define IN_DIM   512
#define D_OUT    512
#define NB       9
#define N_U      64
#define WDIM     128
#define KDIM     (NB * IN_DIM)   // 4608

// GEMM tiling: 128x128 WG tile, 8 wave32 waves -> each wave 32x64 (2x4 WMMA tiles)
#define BM 128
#define BN 128
#define BK 32
#define LDA 40   // bf16 elements per LDS row (80B: 16B-aligned rows, conflict-free b128 reads)
#define LDB 40

__device__ __forceinline__ void split_bf16(float v, __bf16& hi, __bf16& lo) {
    hi = (__bf16)v;                    // RNE truncate to bf16
    lo = (__bf16)(v - (float)hi);      // residual (2-term split: hi*hi + hi*lo + lo*hi ~ fp32)
}

// Assemble a v16bf fragment from two 16-byte LDS chunks (ds_load_b128 each)
__device__ __forceinline__ v16bf frag2(const __bf16* p0, const __bf16* p1) {
    bf16x8 a = *(const bf16x8*)p0;
    bf16x8 b = *(const bf16x8*)p1;
    return __builtin_shufflevector(a, b, 0,1,2,3,4,5,6,7,8,9,10,11,12,13,14,15);
}

// ---------------------------------------------------------------------------
// Kernel 1: B-spline basis (order-4, 12 knots -> 8 bases) + SiLU gate.
// g_ws layout: [batch][NB][N_U]  (19 MB, stays resident in the 192MB L2)
// ---------------------------------------------------------------------------
__global__ void k_spline(const float* __restrict__ x, const float* __restrict__ grid,
                         const int* __restrict__ u, float* __restrict__ g_ws) {
    int tid = blockIdx.x * blockDim.x + threadIdx.x;
    if (tid >= BATCH * N_U) return;
    int ur = tid & (N_U - 1);
    int b  = tid >> 6;
    const float* gr = grid + ur * 6;
    float g0 = gr[0], g5 = gr[5];
    float h = (g5 - g0) * 0.2f;
    float kn[12];
    kn[0] = g0 - 3.f*h; kn[1] = g0 - 2.f*h; kn[2] = g0 - h;
    #pragma unroll
    for (int j = 0; j < 6; ++j) kn[3 + j] = gr[j];
    kn[9] = g5 + h; kn[10] = g5 + 2.f*h; kn[11] = g5 + 3.f*h;

    float t = x[b * GATED_IN + u[ur]];
    float v[11];
    #pragma unroll
    for (int j = 0; j < 11; ++j) v[j] = (t >= kn[j] && t < kn[j + 1]) ? 1.f : 0.f;
    #pragma unroll
    for (int p = 1; p <= 3; ++p)
        for (int j = 0; j < 11 - p; ++j)
            v[j] = (t - kn[j]) / (kn[j + p] - kn[j]) * v[j]
                 + (kn[j + p + 1] - t) / (kn[j + p + 1] - kn[j + 1]) * v[j + 1];

    float* dst = g_ws + (b * NB) * N_U + ur;
    #pragma unroll
    for (int c = 0; c < 8; ++c) dst[c * N_U] = v[c];
    dst[8 * N_U] = t / (1.f + __expf(-t));   // silu
}

// ---------------------------------------------------------------------------
// Kernel 2: lazy_P weight transform. For output row r of the concatenated block
// result: gather s rows of w_r via perm, apply small P (4x4 or 8x8), scatter to
// W[d][c*512+ii] (q = perm[r], d=q>>9, ii=q&511), stored as split bf16 [n][k].
// ---------------------------------------------------------------------------
__global__ void k_prep_w(const float* __restrict__ lw, const float* __restrict__ Pa,
                         const float* __restrict__ Pb8, const int* __restrict__ perm,
                         __bf16* __restrict__ Whi, __bf16* __restrict__ Wlo) {
    int r = blockIdx.x * blockDim.x + threadIdx.x;
    if (r >= D_OUT * IN_DIM) return;
    int s, i, base;
    const float* P;
    if (r < 131072) { s = 4; i = r & 3; base = r - i; P = Pa; }
    else { int r2 = r - 131072; s = 8; i = r2 & 7; base = 131072 + (r2 - i); P = Pb8; }

    float acc[NB];
    #pragma unroll
    for (int c = 0; c < NB; ++c) acc[c] = 0.f;
    for (int j = 0; j < s; ++j) {
        int q = perm[base + j];
        float pij = P[i * s + j];
        const float* wr = lw + (q >> 9) * KDIM + (q & 511);  // w_r[q][c] = lw[d][c*512+ii]
        #pragma unroll
        for (int c = 0; c < NB; ++c) acc[c] += pij * wr[c * IN_DIM];
    }
    int q = perm[r];
    int d = q >> 9, ii = q & 511;
    #pragma unroll
    for (int c = 0; c < NB; ++c) {
        __bf16 hi, lo; split_bf16(acc[c], hi, lo);
        Whi[d * KDIM + c * IN_DIM + ii] = hi;
        Wlo[d * KDIM + c * IN_DIM + ii] = lo;
    }
}

// ---------------------------------------------------------------------------
// Kernel 3: bi_W[j][i] = sum_w bi_w[w] * bi_basis[w][j][i], split bf16 [n=j][k=i]
// ---------------------------------------------------------------------------
__global__ void k_prep_biw(const float* __restrict__ bw, const float* __restrict__ bb,
                           __bf16* __restrict__ Bhi, __bf16* __restrict__ Blo) {
    int t = blockIdx.x * blockDim.x + threadIdx.x;
    if (t >= D_OUT * D_OUT) return;
    int i = t & 511, j = t >> 9;
    float acc = 0.f;
    for (int w = 0; w < WDIM; ++w) acc += bw[w] * bb[(w * D_OUT + j) * D_OUT + i];
    __bf16 hi, lo; split_bf16(acc, hi, lo);
    Bhi[j * D_OUT + i] = hi;
    Blo[j * D_OUT + i] = lo;
}

// Kernel 4: bias = Pb @ linear_bias
__global__ void k_bias(const float* __restrict__ Pb, const float* __restrict__ lb,
                       float* __restrict__ bias) {
    int d = blockIdx.x * blockDim.x + threadIdx.x;
    if (d >= D_OUT) return;
    float acc = 0.f;
    for (int k = 0; k < D_OUT; ++k) acc += Pb[d * D_OUT + k] * lb[k];
    bias[d] = acc;
}

// ---------------------------------------------------------------------------
// Kernel 5: main GEMM. out1[b][d] = sum_k acts[b][k] * W[d][k] + bias[d]
// acts is NEVER materialized: A-tiles built in LDS from g_ws (L2) and x.
// bf16 split: acc += Ah*Bh + Ah*Bl + Al*Bh  (3x v_wmma_f32_16x16x32_bf16)
// ---------------------------------------------------------------------------
__global__ __launch_bounds__(256) void k_gemm1(
    const float* __restrict__ x, const float* __restrict__ g_ws,
    const int* __restrict__ indices,
    const __bf16* __restrict__ Whi, const __bf16* __restrict__ Wlo,
    const float* __restrict__ bias, float* __restrict__ out1) {
    __shared__ __bf16 sAhi[BM * LDA], sAlo[BM * LDA];
    __shared__ __bf16 sBhi[BN * LDB], sBlo[BN * LDB];

    const int tid  = threadIdx.x;
    const int lane = tid & 31, wave = tid >> 5;
    const int wm = wave & 3, wn = wave >> 2;        // 4 waves along M, 2 along N
    const int half = lane >> 4, lrow = lane & 15;   // WMMA lane decomposition
    const int mBase = blockIdx.y * BM;
    const int nBase = blockIdx.x * BN;
    const int arow = tid >> 1, ahalf = tid & 1;     // LDS fill: 2 threads per row

    v8f acc[2][4];
    #pragma unroll
    for (int mt = 0; mt < 2; ++mt)
        #pragma unroll
        for (int nt = 0; nt < 4; ++nt)
            #pragma unroll
            for (int e = 0; e < 8; ++e) acc[mt][nt][e] = 0.f;

    for (int kt = 0; kt < KDIM; kt += BK) {
        __syncthreads();
        // ---- A tile: acts[b][k] = g[b][nb][indices[i]] * x[b][i], k = nb*512+i
        {
            int nb = kt >> 9;                       // BK=32 tile never crosses nb
            int gb = mBase + arow;
            const float* xrow = x + gb * GATED_IN;
            const float* grow = g_ws + (gb * NB + nb) * N_U;
            int i0 = (kt & 511);
            #pragma unroll
            for (int e = 0; e < 16; ++e) {
                int kk = ahalf * 16 + e;
                int i  = i0 + kk;
                float v = grow[indices[i]] * xrow[i];
                __bf16 hi, lo; split_bf16(v, hi, lo);
                sAhi[arow * LDA + kk] = hi;
                sAlo[arow * LDA + kk] = lo;
            }
        }
        // ---- B tile: W split-bf16, global layout [n][k] -> LDS [n][k]
        {
            const __bf16* ph = Whi + (size_t)(nBase + arow) * KDIM + kt + ahalf * 16;
            const __bf16* pl = Wlo + (size_t)(nBase + arow) * KDIM + kt + ahalf * 16;
            *(bf16x8*)&sBhi[arow * LDB + ahalf * 16]     = *(const bf16x8*)ph;
            *(bf16x8*)&sBhi[arow * LDB + ahalf * 16 + 8] = *(const bf16x8*)(ph + 8);
            *(bf16x8*)&sBlo[arow * LDB + ahalf * 16]     = *(const bf16x8*)pl;
            *(bf16x8*)&sBlo[arow * LDB + ahalf * 16 + 8] = *(const bf16x8*)(pl + 8);
        }
        __syncthreads();
        if (kt + BK < KDIM)   // pull next W tile toward L0 (global_prefetch_b8)
            __builtin_prefetch(Whi + (size_t)(nBase + arow) * KDIM + kt + BK, 0, 1);

        // ---- fragments per ISA 7.12.2 16-bit A layout: K = half*8+e (+8 for e>=8)
        v16bf ah[2], al[2];
        #pragma unroll
        for (int mt = 0; mt < 2; ++mt) {
            const __bf16* pa = &sAhi[(wm * 32 + mt * 16 + lrow) * LDA + half * 8];
            ah[mt] = frag2(pa, pa + 16);
            const __bf16* pb = &sAlo[(wm * 32 + mt * 16 + lrow) * LDA + half * 8];
            al[mt] = frag2(pb, pb + 16);
        }
        #pragma unroll
        for (int nt = 0; nt < 4; ++nt) {
            // B layout: N = lane&15, K = half*8 + e (+8 for e>=8), same as A
            const __bf16* pb = &sBhi[(wn * 64 + nt * 16 + lrow) * LDB + half * 8];
            v16bf bh = frag2(pb, pb + 16);
            const __bf16* pl = &sBlo[(wn * 64 + nt * 16 + lrow) * LDB + half * 8];
            v16bf bl = frag2(pl, pl + 16);
            #pragma unroll
            for (int mt = 0; mt < 2; ++mt) {
                acc[mt][nt] = __builtin_amdgcn_wmma_f32_16x16x32_bf16(
                    false, ah[mt], false, bh, (short)0, acc[mt][nt], false, false);
                acc[mt][nt] = __builtin_amdgcn_wmma_f32_16x16x32_bf16(
                    false, ah[mt], false, bl, (short)0, acc[mt][nt], false, false);
                acc[mt][nt] = __builtin_amdgcn_wmma_f32_16x16x32_bf16(
                    false, al[mt], false, bh, (short)0, acc[mt][nt], false, false);
            }
        }
    }
    // ---- epilogue: C/D layout M = e + 8*half, N = lane&15
    #pragma unroll
    for (int mt = 0; mt < 2; ++mt)
        #pragma unroll
        for (int nt = 0; nt < 4; ++nt) {
            int gm0 = mBase + wm * 32 + mt * 16 + half * 8;
            int gn  = nBase + wn * 64 + nt * 16 + lrow;
            float bv = bias[gn];
            #pragma unroll
            for (int e = 0; e < 8; ++e)
                out1[(size_t)(gm0 + e) * D_OUT + gn] = acc[mt][nt][e] + bv;
        }
}

// ---------------------------------------------------------------------------
// Kernel 6: out = 0.1*(out1 @ bi_W^T) + out1.  K = 512; same WMMA core.
// ---------------------------------------------------------------------------
__global__ __launch_bounds__(256) void k_gemm2(
    const float* __restrict__ out1, const __bf16* __restrict__ Bhi,
    const __bf16* __restrict__ Blo, float* __restrict__ dout) {
    __shared__ __bf16 sAhi[BM * LDA], sAlo[BM * LDA];
    __shared__ __bf16 sBhi[BN * LDB], sBlo[BN * LDB];

    const int tid  = threadIdx.x;
    const int lane = tid & 31, wave = tid >> 5;
    const int wm = wave & 3, wn = wave >> 2;
    const int half = lane >> 4, lrow = lane & 15;
    const int mBase = blockIdx.y * BM;
    const int nBase = blockIdx.x * BN;
    const int arow = tid >> 1, ahalf = tid & 1;

    v8f acc[2][4];
    #pragma unroll
    for (int mt = 0; mt < 2; ++mt)
        #pragma unroll
        for (int nt = 0; nt < 4; ++nt)
            #pragma unroll
            for (int e = 0; e < 8; ++e) acc[mt][nt][e] = 0.f;

    for (int kt = 0; kt < D_OUT; kt += BK) {
        __syncthreads();
        {
            const float* src = out1 + (size_t)(mBase + arow) * D_OUT + kt + ahalf * 16;
            #pragma unroll
            for (int e = 0; e < 16; ++e) {
                __bf16 hi, lo; split_bf16(src[e], hi, lo);
                sAhi[arow * LDA + ahalf * 16 + e] = hi;
                sAlo[arow * LDA + ahalf * 16 + e] = lo;
            }
        }
        {
            const __bf16* ph = Bhi + (size_t)(nBase + arow) * D_OUT + kt + ahalf * 16;
            const __bf16* pl = Blo + (size_t)(nBase + arow) * D_OUT + kt + ahalf * 16;
            *(bf16x8*)&sBhi[arow * LDB + ahalf * 16]     = *(const bf16x8*)ph;
            *(bf16x8*)&sBhi[arow * LDB + ahalf * 16 + 8] = *(const bf16x8*)(ph + 8);
            *(bf16x8*)&sBlo[arow * LDB + ahalf * 16]     = *(const bf16x8*)pl;
            *(bf16x8*)&sBlo[arow * LDB + ahalf * 16 + 8] = *(const bf16x8*)(pl + 8);
        }
        __syncthreads();

        v16bf ah[2], al[2];
        #pragma unroll
        for (int mt = 0; mt < 2; ++mt) {
            const __bf16* pa = &sAhi[(wm * 32 + mt * 16 + lrow) * LDA + half * 8];
            ah[mt] = frag2(pa, pa + 16);
            const __bf16* pb = &sAlo[(wm * 32 + mt * 16 + lrow) * LDA + half * 8];
            al[mt] = frag2(pb, pb + 16);
        }
        #pragma unroll
        for (int nt = 0; nt < 4; ++nt) {
            const __bf16* pb = &sBhi[(wn * 64 + nt * 16 + lrow) * LDB + half * 8];
            v16bf bh = frag2(pb, pb + 16);
            const __bf16* pl = &sBlo[(wn * 64 + nt * 16 + lrow) * LDB + half * 8];
            v16bf bl = frag2(pl, pl + 16);
            #pragma unroll
            for (int mt = 0; mt < 2; ++mt) {
                acc[mt][nt] = __builtin_amdgcn_wmma_f32_16x16x32_bf16(
                    false, ah[mt], false, bh, (short)0, acc[mt][nt], false, false);
                acc[mt][nt] = __builtin_amdgcn_wmma_f32_16x16x32_bf16(
                    false, ah[mt], false, bl, (short)0, acc[mt][nt], false, false);
                acc[mt][nt] = __builtin_amdgcn_wmma_f32_16x16x32_bf16(
                    false, al[mt], false, bh, (short)0, acc[mt][nt], false, false);
            }
        }
    }
    #pragma unroll
    for (int mt = 0; mt < 2; ++mt)
        #pragma unroll
        for (int nt = 0; nt < 4; ++nt) {
            int gm0 = mBase + wm * 32 + mt * 16 + half * 8;
            int gn  = nBase + wn * 64 + nt * 16 + lrow;
            #pragma unroll
            for (int e = 0; e < 8; ++e) {
                size_t o = (size_t)(gm0 + e) * D_OUT + gn;
                dout[o] = 0.1f * acc[mt][nt][e] + out1[o];
            }
        }
}

// ---------------------------------------------------------------------------
extern "C" void kernel_launch(void* const* d_in, const int* in_sizes, int n_in,
                              void* d_out, int out_size, void* d_ws, size_t ws_size,
                              hipStream_t stream) {
    const float* x       = (const float*)d_in[0];
    const float* grid    = (const float*)d_in[1];
    const float* lw      = (const float*)d_in[2];
    const float* lb      = (const float*)d_in[3];
    const float* Pa      = (const float*)d_in[4];
    const float* Pb8     = (const float*)d_in[5];
    const float* Pbm     = (const float*)d_in[6];
    const float* bib     = (const float*)d_in[7];
    const float* biw     = (const float*)d_in[8];
    const int*   u       = (const int*)d_in[9];
    const int*   indices = (const int*)d_in[10];
    const int*   perm    = (const int*)d_in[11];

    float*  g_ws = (float*)d_ws;                       // 8192*9*64
    float*  out1 = g_ws + (size_t)BATCH * NB * N_U;    // 8192*512
    float*  bias = out1 + (size_t)BATCH * D_OUT;       // 512
    __bf16* Whi  = (__bf16*)(bias + D_OUT);            // 512*4608
    __bf16* Wlo  = Whi + (size_t)D_OUT * KDIM;
    __bf16* BWhi = Wlo + (size_t)D_OUT * KDIM;         // 512*512
    __bf16* BWlo = BWhi + (size_t)D_OUT * D_OUT;
    float*  out  = (float*)d_out;

    k_spline  <<<(BATCH * N_U) / 256, 256, 0, stream>>>(x, grid, u, g_ws);
    k_prep_w  <<<(D_OUT * IN_DIM) / 256, 256, 0, stream>>>(lw, Pa, Pb8, perm, Whi, Wlo);
    k_prep_biw<<<(D_OUT * D_OUT) / 256, 256, 0, stream>>>(biw, bib, BWhi, BWlo);
    k_bias    <<<2, 256, 0, stream>>>(Pbm, lb, bias);

    dim3 gg(D_OUT / BN, BATCH / BM);   // 4 x 64 = 256 workgroups
    k_gemm1<<<gg, 256, 0, stream>>>(x, g_ws, indices, Whi, Wlo, bias, out1);
    k_gemm2<<<gg, 256, 0, stream>>>(out1, BWhi, BWlo, out);
}
